// MotionEncoder_36309653520631
// MI455X (gfx1250) — matrix operand order repaired and checked by reading, not compiled, not run on hardware
//
#include <hip/hip_runtime.h>

// MotionEncoder: 43 per-joint MLPs (d in {3,2,1} -> 256 -> 256), gelu(tanh) between.
// Layer 2 (35 GFLOP of GEMM) runs on v_wmma_f32_16x16x32_bf16; layer 1 is fused VALU.

#define LAT   256
#define BT    6272          // 32*196 rows
#define MT    64            // rows per workgroup
#define KC    32            // K chunk per WMMA step
#define HSTR  (LAT + 8)     // padded Ht row stride (bank-conflict-free frag loads)
#define WSTR  (KC + 8)      // padded Wt row stride
#define NJALL 43
#define XF    79            // flat feature width of x

typedef __attribute__((ext_vector_type(16))) __bf16       v16bf;
typedef __attribute__((ext_vector_type(8)))  float        v8f;
typedef __attribute__((ext_vector_type(4)))  unsigned int u32x4;

struct Frag { u32x4 lo, hi; };   // 32 bytes == v16bf

__device__ __forceinline__ unsigned short f2bf(float f) {
  unsigned int u = __builtin_bit_cast(unsigned int, f);
  u += 0x7FFFu + ((u >> 16) & 1u);            // round-to-nearest-even
  return (unsigned short)(u >> 16);
}

__device__ __forceinline__ float gelu_tanh(float x) {
  const float k0 = 0.7978845608028654f;       // sqrt(2/pi)
  const float k1 = 0.044715f;
  float inner = k0 * (x + k1 * x * x * x);
  return 0.5f * x * (1.0f + tanhf(inner));
}

__global__ __launch_bounds__(256)
void motion_encoder_kernel(const float* __restrict__ x,
                           const float* __restrict__ W1_3, const float* __restrict__ b1_3,
                           const float* __restrict__ W2_3, const float* __restrict__ b2_3,
                           const float* __restrict__ W1_2, const float* __restrict__ b1_2,
                           const float* __restrict__ W2_2, const float* __restrict__ b2_2,
                           const float* __restrict__ W1_1, const float* __restrict__ b1_1,
                           const float* __restrict__ W2_1, const float* __restrict__ b2_1,
                           float* __restrict__ out)
{
  __shared__ __attribute__((aligned(16))) unsigned short Ht[MT * HSTR];   // gelu(h) bf16 [row][l]
  __shared__ __attribute__((aligned(16))) unsigned short Wt[LAT * WSTR];  // W2^T chunk bf16 [n][k]
  __shared__ __attribute__((aligned(16))) float          xs[MT * 3];

  const int t    = threadIdx.x;
  const int rt   = blockIdx.x;          // row tile: 0..97
  const int j    = blockIdx.y;          // joint:    0..42
  const int row0 = rt * MT;

  // ---- per-joint parameter selection (groups are contiguous in the flat axis)
  const float *W1, *b1, *W2, *b2;
  int d, xoff;
  if (j < 13)      { int jj = j;      d = 3; xoff = jj * 3;
                     W1 = W1_3 + jj*3*LAT; b1 = b1_3 + jj*LAT; W2 = W2_3 + jj*LAT*LAT; b2 = b2_3 + jj*LAT; }
  else if (j < 23) { int jj = j - 13; d = 2; xoff = 39 + jj * 2;
                     W1 = W1_2 + jj*2*LAT; b1 = b1_2 + jj*LAT; W2 = W2_2 + jj*LAT*LAT; b2 = b2_2 + jj*LAT; }
  else             { int jj = j - 23; d = 1; xoff = 59 + jj;
                     W1 = W1_1 + jj*LAT;   b1 = b1_1 + jj*LAT; W2 = W2_1 + jj*LAT*LAT; b2 = b2_1 + jj*LAT; }

  // ---- stage x tile
  for (int i = t; i < MT * d; i += 256) {
    int r = i / d, dd = i - r * d;
    xs[i] = x[(row0 + r) * XF + xoff + dd];
  }
  __syncthreads();

  // ---- layer 1 + gelu -> Ht (bf16), each thread owns an l-pair over 32 rows
  {
    const int l0    = (t & 127) * 2;
    const int rbase = (t >> 7) * (MT / 2);
    float w1v[3][2];
    const float b10 = b1[l0], b11 = b1[l0 + 1];
    for (int dd = 0; dd < d; ++dd) {
      w1v[dd][0] = W1[dd * LAT + l0];
      w1v[dd][1] = W1[dd * LAT + l0 + 1];
    }
    for (int r = rbase; r < rbase + MT / 2; ++r) {
      float a0 = b10, a1 = b11;
      for (int dd = 0; dd < d; ++dd) {
        float xv = xs[r * d + dd];
        a0 += xv * w1v[dd][0];
        a1 += xv * w1v[dd][1];
      }
      unsigned int packed = (unsigned int)f2bf(gelu_tanh(a0))
                          | ((unsigned int)f2bf(gelu_tanh(a1)) << 16);
      *reinterpret_cast<unsigned int*>(&Ht[r * HSTR + l0]) = packed;
    }
  }

  // ---- wave decomposition: waves 0..3 -> M rows, wave>>2 -> N half
  const int wave  = t >> 5;
  const int lane  = t & 31;
  const int lh    = lane >> 4;          // lane[4]
  const int l15   = lane & 15;          // lane[3:0]
  const int mrow  = (wave & 3) * 16;
  const int nbase = (wave >> 2) * 128;

  v8f acc[8];
  for (int nt = 0; nt < 8; ++nt) {
    float bv = b2[nbase + nt * 16 + l15];     // C/D: N fixed per lane -> bias splat
    for (int e = 0; e < 8; ++e) acc[nt][e] = bv;
  }

  // ---- K loop: stage W2^T chunk -> 8x v_wmma_f32_16x16x32_bf16 per wave
  for (int kc = 0; kc < 8; ++kc) {
    const int kbase = kc * KC;
    __syncthreads();                           // prior chunk's Wt reads done
    for (int i = t; i < LAT * KC; i += 256) {  // thread t owns n = t, all kk
      int kk = i >> 8;
      int n  = i & 255;
      Wt[n * WSTR + kk] = f2bf(W2[(kbase + kk) * LAT + n]);  // coalesced read, transposed write
    }
    __syncthreads();

    // A fragment: 16x32 bf16  (lane -> M, K runs {base, base+16} of 8, base = lh*8)
    Frag fa;
    const unsigned short* hrow = &Ht[(mrow + l15) * HSTR + kbase + lh * 8];
    fa.lo = *reinterpret_cast<const u32x4*>(hrow);
    fa.hi = *reinterpret_cast<const u32x4*>(hrow + 16);
    v16bf A = __builtin_bit_cast(v16bf, fa);

    for (int nt = 0; nt < 8; ++nt) {
      // B fragment: 32x16 bf16 (lane -> N, K runs {base, base+16} of 8, base = lh*8)
      Frag fb;
      const unsigned short* wrow = &Wt[(nbase + nt * 16 + l15) * WSTR + lh * 8];
      fb.lo = *reinterpret_cast<const u32x4*>(wrow);
      fb.hi = *reinterpret_cast<const u32x4*>(wrow + 16);
      v16bf B = __builtin_bit_cast(v16bf, fb);
      acc[nt] = __builtin_amdgcn_wmma_f32_16x16x32_bf16(
          false, A, false, B, (short)0, acc[nt], false, false);
    }
  }

  // ---- epilogue: out[row][j][m], C/D layout M = e + 8*lh, N = l15
  for (int nt = 0; nt < 8; ++nt) {
    const int m = nbase + nt * 16 + l15;
    for (int e = 0; e < 8; ++e) {
      const int row = row0 + mrow + lh * 8 + e;
      out[((size_t)row * NJALL + j) * LAT + m] = acc[nt][e];
    }
  }
}

extern "C" void kernel_launch(void* const* d_in, const int* in_sizes, int n_in,
                              void* d_out, int out_size, void* d_ws, size_t ws_size,
                              hipStream_t stream) {
  const float* x    = (const float*)d_in[0];
  const float* W1_3 = (const float*)d_in[1];
  const float* b1_3 = (const float*)d_in[2];
  const float* W2_3 = (const float*)d_in[3];
  const float* b2_3 = (const float*)d_in[4];
  const float* W1_2 = (const float*)d_in[5];
  const float* b1_2 = (const float*)d_in[6];
  const float* W2_2 = (const float*)d_in[7];
  const float* b2_2 = (const float*)d_in[8];
  const float* W1_1 = (const float*)d_in[9];
  const float* b1_1 = (const float*)d_in[10];
  const float* W2_1 = (const float*)d_in[11];
  const float* b2_1 = (const float*)d_in[12];
  float* out = (float*)d_out;

  dim3 grid(BT / MT, NJALL, 1);   // 98 x 43
  motion_encoder_kernel<<<grid, 256, 0, stream>>>(
      x, W1_3, b1_3, W2_3, b2_3, W1_2, b1_2, W2_2, b2_2, W1_1, b1_1, W2_1, b2_1, out);
}
